// HybridFFN_34557306864165
// MI455X (gfx1250) — compile-verified, matrix operand order checked by reading, not run on hardware
//
#include <hip/hip_runtime.h>
#include <math.h>

// Problem constants (B=4, S=2048, D=1024, INNER=4096, E=4, K=1, ALPHA=0.5)
#define TOKENS 8192
#define DMODEL 1024
#define INNER  4096
#define NEXP   4
#define ALPHA_ 0.5f

#define BM 128
#define BN 128      // dense1/dense2/moe2 block tile N (waves 2x4, each 64x32)
#define BNM 64      // moe1 block tile N (waves 4x2, each 32x32, dual acc)
#define BK 32
#define PITCH 40    // halves per LDS row: 32 data + 8 pad -> conflict-free b128 frags
#define PADMAX (TOKENS + NEXP * BM)   // 8704 padded gathered rows

typedef _Float16 half_t;
typedef __attribute__((ext_vector_type(16))) _Float16 v16h;
typedef __attribute__((ext_vector_type(8)))  _Float16 v8h;
typedef __attribute__((ext_vector_type(8)))  float    v8f;

struct __attribute__((aligned(8))) H4 { half_t a, b, c, d; };

// ---------- CDNA5 async copy: global -> LDS, 16B per lane ----------
__device__ __forceinline__ void async_b128(uint32_t ldsByteAddr, const half_t* gptr) {
  asm volatile("global_load_async_to_lds_b128 %0, %1, off"
               :: "v"(ldsByteAddr), "v"((uint64_t)(uintptr_t)gptr)
               : "memory");
}
__device__ __forceinline__ void wait_async0() {
  asm volatile("s_wait_asynccnt 0x0" ::: "memory");
}

// ---------- WMMA fragment load from LDS (ISA 7.12.2 16-bit layouts) ----------
__device__ __forceinline__ v16h frag_load(const half_t* lds, int rowBase, int lane) {
  int r  = rowBase + (lane & 15);
  int kb = (lane >> 4) << 3;
  const half_t* p = lds + r * PITCH + kb;   // 16B aligned
  v8h lo = *(const v8h*)(p);
  v8h hi = *(const v8h*)(p + 16);
  v16h f;
#pragma unroll
  for (int i = 0; i < 8; ++i) { f[i] = lo[i]; f[i + 8] = hi[i]; }
  return f;
}

__device__ __forceinline__ v8f wmma_f16(v16h a, v16h b, v8f c) {
  return __builtin_amdgcn_wmma_f32_16x16x32_f16(false, a, false, b, (short)0, c,
                                                false, false);
}

// ---------- async tile loaders (f16 row-major src, ROWS x 32 halves) ----------
__device__ __forceinline__ void async_tile128(uint32_t ldsBase, const half_t* src,
                                              int ldk, int rowStart, int k0,
                                              const int* perm, int tid) {
#pragma unroll
  for (int j = 0; j < 2; ++j) {
    int idx = tid + j * 256;         // 128 rows * 4 chunks
    int row = idx >> 2, c = idx & 3;
    int r = rowStart + row;
    if (perm) { int t = perm[r]; r = (t < 0) ? 0 : t; }  // pad rows: data discarded later
    async_b128(ldsBase + (uint32_t)(row * PITCH + c * 8) * 2,
               src + (size_t)r * ldk + k0 + c * 8);
  }
}
__device__ __forceinline__ void async_tile64(uint32_t ldsBase, const half_t* src,
                                             int ldk, int rowStart, int k0, int tid) {
  int row = tid >> 2, c = tid & 3;   // 64 rows * 4 chunks
  async_b128(ldsBase + (uint32_t)(row * PITCH + c * 8) * 2,
             src + (size_t)(rowStart + row) * ldk + k0 + c * 8);
}

// ================= pre-pass: fp32 -> f16 =================
__global__ __launch_bounds__(256)
void cvt_f32_f16_kernel(const float* __restrict__ in, half_t* __restrict__ out,
                        int n4) {
  int i = blockIdx.x * 256 + threadIdx.x;
  if (i >= n4) return;
  float4 v = *(const float4*)(in + (size_t)i * 4);
  H4 h; h.a = (half_t)v.x; h.b = (half_t)v.y; h.c = (half_t)v.z; h.d = (half_t)v.w;
  *(H4*)(out + (size_t)i * 4) = h;
}

// in: [K][N] f32 row-major -> out: [N][K] f16 (per blockIdx.z matrix)
__global__ __launch_bounds__(256)
void transpose_f32_f16_kernel(const float* __restrict__ in,
                              half_t* __restrict__ out, int K, int N) {
  __shared__ float tile[32][33];
  size_t zo = (size_t)blockIdx.z * K * N;
  int k0 = blockIdx.y * 32, n0 = blockIdx.x * 32;
  int tx = threadIdx.x & 31, ty = threadIdx.x >> 5;
#pragma unroll
  for (int i = 0; i < 4; ++i)
    tile[ty + 8 * i][tx] = in[zo + (size_t)(k0 + ty + 8 * i) * N + (n0 + tx)];
  __syncthreads();
#pragma unroll
  for (int i = 0; i < 4; ++i)
    out[zo + (size_t)(n0 + ty + 8 * i) * K + (k0 + tx)] =
        (half_t)tile[tx][ty + 8 * i];
}

// ====== 128x128 GEMM core (waves 2x4; wave = 64x32 = 4x2 subtiles) ======
// Pipeline per iter: wait(prev issue) -> barrier(publish) -> issue k+1 -> compute k
#define GEMM128_LOOP(SRC_A, LDK_A, PERM_A, SRC_B, LDK_B, KTOT)                  \
  async_tile128(aB, SRC_A, LDK_A, rowStart, 0, PERM_A, tid);                    \
  async_tile128(bB, SRC_B, LDK_B, n0, 0, nullptr, tid);                         \
  const int NT = (KTOT) / BK;                                                   \
  for (int kt = 0; kt < NT; ++kt) {                                             \
    int cur = kt & 1;                                                           \
    wait_async0();                                                              \
    __syncthreads();                                                            \
    if (kt + 1 < NT) {                                                          \
      async_tile128(aB + (cur ^ 1) * TSZ, SRC_A, LDK_A, rowStart,               \
                    (kt + 1) * BK, PERM_A, tid);                                \
      async_tile128(bB + (cur ^ 1) * TSZ, SRC_B, LDK_B, n0, (kt + 1) * BK,      \
                    nullptr, tid);                                              \
    }                                                                           \
    const half_t* cA = ldsA + cur * BM * PITCH;                                 \
    const half_t* cB = ldsB + cur * BN * PITCH;                                 \
    v16h af[4], bf[2];                                                          \
    _Pragma("unroll")                                                           \
    for (int i = 0; i < 4; ++i) af[i] = frag_load(cA, wm * 64 + i * 16, lane);  \
    _Pragma("unroll")                                                           \
    for (int j = 0; j < 2; ++j) bf[j] = frag_load(cB, wn * 32 + j * 16, lane);  \
    _Pragma("unroll")                                                           \
    for (int i = 0; i < 4; ++i)                                                 \
      _Pragma("unroll")                                                         \
      for (int j = 0; j < 2; ++j) acc[i][j] = wmma_f16(af[i], bf[j], acc[i][j]);\
  }

// ================= Dense branch =================
// H = GELU_exact(x @ W1 + b1) f16; W1t: [INNER][DMODEL]
__global__ __launch_bounds__(256)
void dense1_kernel(const half_t* __restrict__ Xh, const half_t* __restrict__ W1t,
                   const float* __restrict__ b1, half_t* __restrict__ H) {
  __shared__ half_t ldsA[2 * BM * PITCH];
  __shared__ half_t ldsB[2 * BN * PITCH];
  int tid = threadIdx.x, lane = tid & 31, wid = tid >> 5;
  int wm = wid >> 2, wn = wid & 3;
  int rowStart = blockIdx.y * BM, n0 = blockIdx.x * BN;
  uint32_t aB = (uint32_t)(uintptr_t)ldsA, bB = (uint32_t)(uintptr_t)ldsB;
  const uint32_t TSZ = BM * PITCH * 2;
  v8f acc[4][2] = {};
  GEMM128_LOOP(Xh, DMODEL, nullptr, W1t, DMODEL, DMODEL)
  int mOff = rowStart + wm * 64 + 8 * (lane >> 4);
  int nOff = n0 + wn * 32 + (lane & 15);
#pragma unroll
  for (int sm = 0; sm < 4; ++sm)
#pragma unroll
    for (int sn = 0; sn < 2; ++sn) {
      int nn = nOff + sn * 16;
      float bias = b1[nn];
#pragma unroll
      for (int r = 0; r < 8; ++r) {
        float v = acc[sm][sn][r] + bias;
        float g = 0.5f * v * (1.0f + erff(v * 0.70710678118654752f));
        H[(size_t)(mOff + sm * 16 + r) * INNER + nn] = (half_t)g;
      }
    }
}

// out = ALPHA*(H @ W2 + b2); W2t: [DMODEL][INNER]
__global__ __launch_bounds__(256)
void dense2_kernel(const half_t* __restrict__ H, const half_t* __restrict__ W2t,
                   const float* __restrict__ b2, float* __restrict__ OUT) {
  __shared__ half_t ldsA[2 * BM * PITCH];
  __shared__ half_t ldsB[2 * BN * PITCH];
  int tid = threadIdx.x, lane = tid & 31, wid = tid >> 5;
  int wm = wid >> 2, wn = wid & 3;
  int rowStart = blockIdx.y * BM, n0 = blockIdx.x * BN;
  uint32_t aB = (uint32_t)(uintptr_t)ldsA, bB = (uint32_t)(uintptr_t)ldsB;
  const uint32_t TSZ = BM * PITCH * 2;
  v8f acc[4][2] = {};
  GEMM128_LOOP(H, INNER, nullptr, W2t, INNER, INNER)
  int mOff = rowStart + wm * 64 + 8 * (lane >> 4);
  int nOff = n0 + wn * 32 + (lane & 15);
#pragma unroll
  for (int sm = 0; sm < 4; ++sm)
#pragma unroll
    for (int sn = 0; sn < 2; ++sn) {
      int nn = nOff + sn * 16;
      float bias = b2[nn];
#pragma unroll
      for (int r = 0; r < 8; ++r)
        OUT[(size_t)(mOff + sm * 16 + r) * DMODEL + nn] =
            ALPHA_ * (acc[sm][sn][r] + bias);
    }
}

// out[perm[g]] += (1-ALPHA)*(Hm[g] @ ew2[e]); e2t: [E][DMODEL][INNER]
__global__ __launch_bounds__(256)
void moe2_kernel(const half_t* __restrict__ H, const half_t* __restrict__ E2t,
                 const int* __restrict__ perm, const int* __restrict__ po,
                 float* __restrict__ OUT) {
  __shared__ half_t ldsA[2 * BM * PITCH];
  __shared__ half_t ldsB[2 * BN * PITCH];
  int rowStart = blockIdx.y * BM;
  if (rowStart >= po[4]) return;               // uniform
  int e = 0;
  while (e < 3 && rowStart >= po[e + 1]) ++e;  // BM-aligned segments
  const half_t* W2e = E2t + (size_t)e * DMODEL * INNER;
  int tid = threadIdx.x, lane = tid & 31, wid = tid >> 5;
  int wm = wid >> 2, wn = wid & 3;
  int n0 = blockIdx.x * BN;
  uint32_t aB = (uint32_t)(uintptr_t)ldsA, bB = (uint32_t)(uintptr_t)ldsB;
  const uint32_t TSZ = BM * PITCH * 2;
  v8f acc[4][2] = {};
  GEMM128_LOOP(H, INNER, nullptr, W2e, INNER, INNER)
  int mOff = rowStart + wm * 64 + 8 * (lane >> 4);
  int nOff = n0 + wn * 32 + (lane & 15);
#pragma unroll
  for (int sm = 0; sm < 4; ++sm)
#pragma unroll
    for (int sn = 0; sn < 2; ++sn) {
      int nn = nOff + sn * 16;
#pragma unroll
      for (int r = 0; r < 8; ++r) {
        int g = mOff + sm * 16 + r;
        int t = perm[g];
        if (t >= 0)
          OUT[(size_t)t * DMODEL + nn] += (1.0f - ALPHA_) * acc[sm][sn][r];
      }
    }
}

// Hm[g] = silu(x[perm[g]] @ ew1[e]) * (x[perm[g]] @ ew3[e]); 128x64 tile, dual acc
__global__ __launch_bounds__(256)
void moe1_kernel(const half_t* __restrict__ Xh, const half_t* __restrict__ E1t,
                 const half_t* __restrict__ E3t, const int* __restrict__ perm,
                 const int* __restrict__ po, half_t* __restrict__ H) {
  __shared__ half_t ldsA[2 * BM * PITCH];
  __shared__ half_t ldsB1[2 * BNM * PITCH];
  __shared__ half_t ldsB3[2 * BNM * PITCH];
  int rowStart = blockIdx.y * BM;
  if (rowStart >= po[4]) return;
  int e = 0;
  while (e < 3 && rowStart >= po[e + 1]) ++e;
  const half_t* W1e = E1t + (size_t)e * INNER * DMODEL;
  const half_t* W3e = E3t + (size_t)e * INNER * DMODEL;
  int tid = threadIdx.x, lane = tid & 31, wid = tid >> 5;
  int wm = wid >> 1, wn = wid & 1;
  int n0 = blockIdx.x * BNM;
  uint32_t aB = (uint32_t)(uintptr_t)ldsA;
  uint32_t b1B = (uint32_t)(uintptr_t)ldsB1, b3B = (uint32_t)(uintptr_t)ldsB3;
  const uint32_t ASZ = BM * PITCH * 2, BSZ = BNM * PITCH * 2;
  v8f acc1[2][2] = {}, acc3[2][2] = {};
  async_tile128(aB, Xh, DMODEL, rowStart, 0, perm, tid);
  async_tile64(b1B, W1e, DMODEL, n0, 0, tid);
  async_tile64(b3B, W3e, DMODEL, n0, 0, tid);
  const int NT = DMODEL / BK;
  for (int kt = 0; kt < NT; ++kt) {
    int cur = kt & 1;
    wait_async0();
    __syncthreads();
    if (kt + 1 < NT) {
      async_tile128(aB + (cur ^ 1) * ASZ, Xh, DMODEL, rowStart, (kt + 1) * BK,
                    perm, tid);
      async_tile64(b1B + (cur ^ 1) * BSZ, W1e, DMODEL, n0, (kt + 1) * BK, tid);
      async_tile64(b3B + (cur ^ 1) * BSZ, W3e, DMODEL, n0, (kt + 1) * BK, tid);
    }
    const half_t* cA = ldsA + cur * BM * PITCH;
    const half_t* c1 = ldsB1 + cur * BNM * PITCH;
    const half_t* c3 = ldsB3 + cur * BNM * PITCH;
    v16h a0 = frag_load(cA, wm * 32, lane);
    v16h a1 = frag_load(cA, wm * 32 + 16, lane);
    v16h p0 = frag_load(c1, wn * 32, lane);
    v16h p1 = frag_load(c1, wn * 32 + 16, lane);
    v16h q0 = frag_load(c3, wn * 32, lane);
    v16h q1 = frag_load(c3, wn * 32 + 16, lane);
    acc1[0][0] = wmma_f16(a0, p0, acc1[0][0]);
    acc1[0][1] = wmma_f16(a0, p1, acc1[0][1]);
    acc1[1][0] = wmma_f16(a1, p0, acc1[1][0]);
    acc1[1][1] = wmma_f16(a1, p1, acc1[1][1]);
    acc3[0][0] = wmma_f16(a0, q0, acc3[0][0]);
    acc3[0][1] = wmma_f16(a0, q1, acc3[0][1]);
    acc3[1][0] = wmma_f16(a1, q0, acc3[1][0]);
    acc3[1][1] = wmma_f16(a1, q1, acc3[1][1]);
  }
  int mOff = rowStart + wm * 32 + 8 * (lane >> 4);
  int nOff = n0 + wn * 32 + (lane & 15);
#pragma unroll
  for (int sm = 0; sm < 2; ++sm)
#pragma unroll
    for (int sn = 0; sn < 2; ++sn) {
      int nn = nOff + sn * 16;
#pragma unroll
      for (int r = 0; r < 8; ++r) {
        float u = acc1[sm][sn][r];
        float w = acc3[sm][sn][r];
        float h = (u / (1.0f + expf(-u))) * w;
        H[(size_t)(mOff + sm * 16 + r) * INNER + nn] = (half_t)h;
      }
    }
}

// ================= Routing =================
__global__ __launch_bounds__(256)
void gate_kernel(const float* __restrict__ X, const float* __restrict__ Wg,
                 int* __restrict__ topi, float* __restrict__ stats) {
  int lane = threadIdx.x & 31, wid = threadIdx.x >> 5;
  int t = blockIdx.x * 8 + wid;
  const float* xr = X + (size_t)t * DMODEL;
  float s0 = 0.f, s1 = 0.f, s2 = 0.f, s3 = 0.f;
  for (int d = lane; d < DMODEL; d += 32) {
    float xv = xr[d];
    float4 w = *(const float4*)(Wg + d * 4);
    s0 += xv * w.x; s1 += xv * w.y; s2 += xv * w.z; s3 += xv * w.w;
  }
#pragma unroll
  for (int off = 16; off > 0; off >>= 1) {
    s0 += __shfl_down(s0, off);
    s1 += __shfl_down(s1, off);
    s2 += __shfl_down(s2, off);
    s3 += __shfl_down(s3, off);
  }
  if (lane == 0) {
    float l[4] = {s0, s1, s2, s3};
    float m = l[0]; int arg = 0;
    for (int e = 1; e < 4; ++e) if (l[e] > m) { m = l[e]; arg = e; }
    float p[4], sum = 0.f;
    for (int e = 0; e < 4; ++e) { p[e] = expf(l[e] - m); sum += p[e]; }
    for (int e = 0; e < 4; ++e) atomicAdd(&stats[e], p[e] / sum);
    atomicAdd(&stats[4 + arg], 1.0f);
    topi[t] = arg;
  }
}

__global__ void zero_stats_kernel(float* stats) {
  if (threadIdx.x < 8) stats[threadIdx.x] = 0.f;
}

__global__ void scan_kernel(const float* __restrict__ stats, int* __restrict__ po,
                            int* __restrict__ cur, float* __restrict__ auxOut) {
  if (threadIdx.x == 0 && blockIdx.x == 0) {
    int off = 0; po[0] = 0;
    float aux = 0.f;
    for (int e = 0; e < NEXP; ++e) {
      int c = (int)(stats[4 + e] + 0.5f);
      aux += ((float)c / (float)TOKENS) * (stats[e] / (float)TOKENS);
      off += ((c + BM - 1) / BM) * BM;
      po[e + 1] = off;
      cur[e] = 0;
    }
    *auxOut = (float)NEXP * aux;
  }
}

__global__ void fill_perm_kernel(int* __restrict__ perm) {
  int i = blockIdx.x * 256 + threadIdx.x;
  if (i < PADMAX) perm[i] = -1;
}

__global__ void scatter_kernel(const int* __restrict__ topi,
                               const int* __restrict__ po, int* __restrict__ cur,
                               int* __restrict__ perm) {
  int t = blockIdx.x * 256 + threadIdx.x;
  if (t >= TOKENS) return;
  int e = topi[t];
  int pos = po[e] + atomicAdd(&cur[e], 1);
  perm[pos] = t;
}

extern "C" void kernel_launch(void* const* d_in, const int* in_sizes, int n_in,
                              void* d_out, int out_size, void* d_ws, size_t ws_size,
                              hipStream_t stream) {
  (void)in_sizes; (void)n_in; (void)out_size; (void)ws_size;
  const float* x   = (const float*)d_in[0];
  const float* W1  = (const float*)d_in[1];
  const float* b1  = (const float*)d_in[2];
  const float* W2  = (const float*)d_in[3];
  const float* b2  = (const float*)d_in[4];
  const float* Wg  = (const float*)d_in[5];
  const float* ew1 = (const float*)d_in[6];
  const float* ew3 = (const float*)d_in[7];
  const float* ew2 = (const float*)d_in[8];
  float* out = (float*)d_out;

  char* ws = (char*)d_ws;
  size_t off = 0;
  half_t* xh  = (half_t*)(ws + off); off += (size_t)TOKENS * DMODEL * 2;
  half_t* W1t = (half_t*)(ws + off); off += (size_t)INNER * DMODEL * 2;
  half_t* W2t = (half_t*)(ws + off); off += (size_t)DMODEL * INNER * 2;
  half_t* e1t = (half_t*)(ws + off); off += (size_t)NEXP * INNER * DMODEL * 2;
  half_t* e3t = (half_t*)(ws + off); off += (size_t)NEXP * INNER * DMODEL * 2;
  half_t* e2t = (half_t*)(ws + off); off += (size_t)NEXP * DMODEL * INNER * 2;
  half_t* H   = (half_t*)(ws + off); off += (size_t)PADMAX * INNER * 2;
  int*   topi  = (int*)(ws + off);   off += (size_t)TOKENS * sizeof(int);
  int*   perm  = (int*)(ws + off);   off += (size_t)PADMAX * sizeof(int);
  float* stats = (float*)(ws + off); off += 8 * sizeof(float);
  int*   po    = (int*)(ws + off);   off += 8 * sizeof(int);
  int*   cur   = (int*)(ws + off);   off += 8 * sizeof(int);

  // --- precision/layout pre-pass: f32 -> f16, weights transposed to [N][K] ---
  cvt_f32_f16_kernel<<<TOKENS * DMODEL / 4 / 256, 256, 0, stream>>>(
      x, xh, TOKENS * DMODEL / 4);
  transpose_f32_f16_kernel<<<dim3(INNER / 32, DMODEL / 32, 1), 256, 0, stream>>>(
      W1, W1t, DMODEL, INNER);
  transpose_f32_f16_kernel<<<dim3(DMODEL / 32, INNER / 32, 1), 256, 0, stream>>>(
      W2, W2t, INNER, DMODEL);
  transpose_f32_f16_kernel<<<dim3(INNER / 32, DMODEL / 32, NEXP), 256, 0, stream>>>(
      ew1, e1t, DMODEL, INNER);
  transpose_f32_f16_kernel<<<dim3(INNER / 32, DMODEL / 32, NEXP), 256, 0, stream>>>(
      ew3, e3t, DMODEL, INNER);
  transpose_f32_f16_kernel<<<dim3(DMODEL / 32, INNER / 32, NEXP), 256, 0, stream>>>(
      ew2, e2t, INNER, DMODEL);

  // --- routing + aux ---
  zero_stats_kernel<<<1, 32, 0, stream>>>(stats);
  gate_kernel<<<TOKENS / 8, 256, 0, stream>>>(x, Wg, topi, stats);
  scan_kernel<<<1, 32, 0, stream>>>(stats, po, cur, out + (size_t)TOKENS * DMODEL);
  fill_perm_kernel<<<(PADMAX + 255) / 256, 256, 0, stream>>>(perm);
  scatter_kernel<<<(TOKENS + 255) / 256, 256, 0, stream>>>(topi, po, cur, perm);

  // --- GEMMs: dense writes out with '='; MoE reuses H then adds ---
  dense1_kernel<<<dim3(INNER / BN, TOKENS / BM), 256, 0, stream>>>(xh, W1t, b1, H);
  dense2_kernel<<<dim3(DMODEL / BN, TOKENS / BM), 256, 0, stream>>>(H, W2t, b2, out);
  moe1_kernel<<<dim3(INNER / BNM, PADMAX / BM), 256, 0, stream>>>(xh, e1t, e3t,
                                                                  perm, po, H);
  moe2_kernel<<<dim3(DMODEL / BN, PADMAX / BM), 256, 0, stream>>>(H, e2t, perm,
                                                                  po, out);
}
